// MoEFFN_15882789061171
// MI455X (gfx1250) — matrix operand order repaired and checked by reading, not compiled, not run on hardware
//
#include <hip/hip_runtime.h>
#include <hip/hip_bf16.h>

#define D_MODEL 512
#define NEXP    8
#define FFN     2048
#define H       1024
#define GRU_OUT 2048
#define BATCH   8
#define TLEN    1024
#define NTOK    (BATCH * TLEN)     /* 8192 */
#define OUTN    (NTOK * D_MODEL)   /* 4194304 */

typedef __attribute__((ext_vector_type(16))) __bf16       v16bf;
typedef __attribute__((ext_vector_type(8)))  float        v8f;
typedef __attribute__((ext_vector_type(4)))  unsigned int v4u;

__device__ __forceinline__ unsigned short f2bf(float x) {
    union { float f; unsigned u; } v; v.f = x;
    unsigned u = v.u;
    u += 0x7fffu + ((u >> 16) & 1u);           // round-to-nearest-even
    return (unsigned short)(u >> 16);
}
__device__ __forceinline__ float bf2f(unsigned short h) {
    union { float f; unsigned u; } v; v.u = ((unsigned)h) << 16;
    return v.f;
}

union FragAB { v16bf v; v4u q[2]; };

// A fragment, 16x32 bf16 row-major. rowptr = this lane's row (M = lane&15).
// lanes 0-15: elems 0..7 = K k0..k0+7, elems 8..15 = K k0+16..k0+23
// lanes 16-31: same shifted by +8.
__device__ __forceinline__ v16bf load_afrag(const unsigned short* rowptr, int k0, int lane) {
    FragAB t;
    const unsigned short* p = rowptr + k0 + ((lane & 16) ? 8 : 0);
    t.q[0] = *(const v4u*)(p);
    t.q[1] = *(const v4u*)(p + 16);
    return t.v;
}
// B fragment, 32x16 bf16. colptr = K-major column for this lane (N = lane&15).
// lanes 0-15: elems 0..7 = K k0..k0+7, elems 8..15 = K k0+16..k0+23; lanes 16-31: same shifted by +8.
__device__ __forceinline__ v16bf load_bfrag(const unsigned short* colptr, int k0, int lane) {
    FragAB t;
    const unsigned short* p = colptr + k0 + ((lane & 16) ? 8 : 0);
    t.q[0] = *(const v4u*)(p);
    t.q[1] = *(const v4u*)(p + 16);
    return t.v;
}

#define WMMA_BF16(A, B, C) \
    __builtin_amdgcn_wmma_f32_16x16x32_bf16(false, (A), false, (B), (short)0, (C), false, false)

// async global -> LDS, 16 bytes per active lane (ASYNCcnt-tracked)
__device__ __forceinline__ void async_g2l_b128(unsigned lds_off, const void* gptr) {
    asm volatile("global_load_async_to_lds_b128 %0, %1, off"
                 :: "v"(lds_off), "v"(gptr) : "memory");
}
__device__ __forceinline__ void wait_async0() {
    asm volatile("s_wait_asynccnt 0" ::: "memory");
}

// ---------------------------------------------------------------- init
__global__ void k_init(float* out, float* ctrl) {
    long i = (long)blockIdx.x * blockDim.x + threadIdx.x;
    long s = (long)gridDim.x * blockDim.x;
    for (long j = i; j < OUTN; j += s) out[j] = 0.f;
    if (i < 64) ctrl[i] = 0.f;  // [0]=lse2, [1..8]=sumP, [9..16]=cnt, [32..39]=ecount(int)
}

// ---------------------------------------------------------------- fp32 -> bf16
__global__ void k_f2bf(const float* __restrict__ src, unsigned short* __restrict__ dst, long n) {
    long i = (long)blockIdx.x * blockDim.x + threadIdx.x;
    long s = (long)gridDim.x * blockDim.x;
    for (long j = i; j < n; j += s) dst[j] = f2bf(src[j]);
}

// [e][r][c] fp32 -> [e][c][r] bf16 (make B operands K-major/contiguous per column)
__global__ void k_t2bf(const float* __restrict__ src, unsigned short* __restrict__ dst,
                       int E, int R, int C) {
    long n = (long)E * R * C;
    long i = (long)blockIdx.x * blockDim.x + threadIdx.x;
    long s = (long)gridDim.x * blockDim.x;
    for (long j = i; j < n; j += s) {
        int c = (int)(j % C);
        long t = j / C;
        int r = (int)(t % R);
        int e = (int)(t / R);
        dst[((long)e * C + c) * R + r] = f2bf(src[j]);
    }
}

// ---------------------------------------------------------------- x @ Wih^T + bih  (both dirs)
// xg layout: [dir][t][b][3H] bf16 (gate order r,z,n)
__global__ void k_xproj(const unsigned short* __restrict__ x_bf,
                        const unsigned short* __restrict__ Wih_bf,
                        const float* __restrict__ bih,
                        unsigned short* __restrict__ xg_bf) {
    int lane = threadIdx.x & 31;
    int wv = (blockIdx.x * blockDim.x + threadIdx.x) >> 5;   // global wave id
    int nt = wv % 192;
    int mt = (wv / 192) % 512;
    int d  = wv / (192 * 512);

    int g = mt * 16 + (lane & 15);               // GEMM row = t*8 + b
    int tt = g >> 3, bb = g & 7;
    const unsigned short* arow = x_bf + ((long)bb * TLEN + tt) * D_MODEL;

    int col = nt * 16 + (lane & 15);
    const unsigned short* bcol = Wih_bf + ((long)d * 3 * H + col) * D_MODEL;
    float bias = bih[d * 3 * H + col];

    v8f c;
#pragma unroll
    for (int j = 0; j < 8; ++j) c[j] = bias;
    for (int k = 0; k < D_MODEL; k += 32) {
        v16bf a = load_afrag(arow, k, lane);
        v16bf b = load_bfrag(bcol, k, lane);
        c = WMMA_BF16(a, b, c);
    }
#pragma unroll
    for (int j = 0; j < 8; ++j) {
        int gr = mt * 16 + j + ((lane & 16) ? 8 : 0);
        int t2 = gr >> 3, b2 = gr & 7;
        xg_bf[(((long)d * TLEN + t2) * BATCH + b2) * (3 * H) + col] = f2bf(c[j]);
    }
}

// ---------------------------------------------------------------- bidirectional GRU scan
// One workgroup (32 waves) per direction; h (16x1024, rows 8..15 zero) lives in LDS bf16.
__global__ void __launch_bounds__(1024) k_gru(const unsigned short* __restrict__ xg_bf,
                                              const unsigned short* __restrict__ Whh_bf,
                                              const float* __restrict__ bhh,
                                              unsigned short* __restrict__ flat_bf) {
    __shared__ unsigned short hbuf[16 * H];       // 32 KB
    int d = blockIdx.x;
    int lane = threadIdx.x & 31;
    int w = threadIdx.x >> 5;                     // 0..31
    for (int i = threadIdx.x; i < 16 * H; i += blockDim.x) hbuf[i] = 0;

    const unsigned short* Wd = Whh_bf + (long)d * 3 * H * H;
    const float* bh = bhh + d * 3 * H;

    v8f hprev[2];
#pragma unroll
    for (int q = 0; q < 2; ++q)
#pragma unroll
        for (int j = 0; j < 8; ++j) hprev[q][j] = 0.f;

    __syncthreads();
    for (int ts = 0; ts < TLEN; ++ts) {
        int t = d ? (TLEN - 1 - ts) : ts;
        const unsigned short* xgrow = xg_bf + ((long)d * TLEN + t) * BATCH * (3 * H);
        float hnew[2][8];

        for (int q = 0; q < 2; ++q) {
            int ci = w + q * 32;                  // hidden chunk 0..63
            int col = ci * 16 + (lane & 15);
            const unsigned short* pr = Wd + (long)(0 * H + col) * H;
            const unsigned short* pz = Wd + (long)(1 * H + col) * H;
            const unsigned short* pn = Wd + (long)(2 * H + col) * H;

            float xr[8], xz[8], xn[8];
            if (lane < 16) {
#pragma unroll
                for (int j = 0; j < 8; ++j) {     // this lane carries batch rows 0..7
                    const unsigned short* xb = xgrow + (long)j * (3 * H);
                    xr[j] = bf2f(xb[col]);
                    xz[j] = bf2f(xb[H + col]);
                    xn[j] = bf2f(xb[2 * H + col]);
                }
            } else {
#pragma unroll
                for (int j = 0; j < 8; ++j) { xr[j] = 0.f; xz[j] = 0.f; xn[j] = 0.f; }
            }
            float br = bh[col], bz = bh[H + col], bn = bh[2 * H + col];
            v8f Cr, Cz, Cn;
#pragma unroll
            for (int j = 0; j < 8; ++j) { Cr[j] = xr[j] + br; Cz[j] = xz[j] + bz; Cn[j] = bn; }

            const unsigned short* hrow = hbuf + (lane & 15) * H;
            for (int k = 0; k < H; k += 32) {
                v16bf a = load_afrag(hrow, k, lane);
                Cr = WMMA_BF16(a, load_bfrag(pr, k, lane), Cr);
                Cz = WMMA_BF16(a, load_bfrag(pz, k, lane), Cz);
                Cn = WMMA_BF16(a, load_bfrag(pn, k, lane), Cn);
            }
#pragma unroll
            for (int j = 0; j < 8; ++j) {         // torch GRU: n = tanh(xn + r*(Whn h + bhn))
                float r = 1.f / (1.f + __expf(-Cr[j]));
                float z = 1.f / (1.f + __expf(-Cz[j]));
                float nn = tanhf(xn[j] + r * Cn[j]);
                hnew[q][j] = (1.f - z) * nn + z * hprev[q][j];
                hprev[q][j] = hnew[q][j];
            }
        }
        __syncthreads();                          // all LDS reads of old h done
        if (lane < 16) {
#pragma unroll
            for (int q = 0; q < 2; ++q) {
                int col = (w + q * 32) * 16 + lane;
#pragma unroll
                for (int j = 0; j < 8; ++j) {
                    float hv = hnew[q][j];
                    hbuf[j * H + col] = f2bf(hv);
                    float a = hv > 0.f ? hv : 0.01f * hv;   // leaky relu, stored activated
                    flat_bf[((long)j * TLEN + t) * GRU_OUT + d * H + col] = f2bf(a);
                }
            }
        }
        __syncthreads();
    }
}

// ---------------------------------------------------------------- gating / routing / aux stats
__global__ void k_gate(const unsigned short* __restrict__ flat_bf,
                       const float* __restrict__ gW, const float* __restrict__ gb,
                       float* ctrl, int* ecount, int* elist, float* wlist) {
    int i = blockIdx.x * blockDim.x + threadIdx.x;
    if (i >= NTOK) return;
    const unsigned short* fr = flat_bf + (long)i * GRU_OUT;
    float acc[NEXP];
#pragma unroll
    for (int e = 0; e < NEXP; ++e) acc[e] = gb[e];
    for (int k = 0; k < GRU_OUT; ++k) {
        float xv = bf2f(fr[k]);
#pragma unroll
        for (int e = 0; e < NEXP; ++e) acc[e] += xv * gW[e * GRU_OUT + k];
    }
    float m = acc[0];
#pragma unroll
    for (int e = 1; e < NEXP; ++e) m = fmaxf(m, acc[e]);
    float s = 0.f, ex[NEXP];
#pragma unroll
    for (int e = 0; e < NEXP; ++e) { ex[e] = __expf(acc[e] - m); s += ex[e]; }
    float lse = m + __logf(s);
    atomicAdd(&ctrl[0], lse * lse);
    float sc[NEXP];
#pragma unroll
    for (int e = 0; e < NEXP; ++e) { sc[e] = ex[e] / s; atomicAdd(&ctrl[1 + e], sc[e]); }
    int i1 = 0;
#pragma unroll
    for (int e = 1; e < NEXP; ++e) if (sc[e] > sc[i1]) i1 = e;
    int i2 = (i1 == 0) ? 1 : 0;
#pragma unroll
    for (int e = 0; e < NEXP; ++e) if (e != i1 && sc[e] > sc[i2]) i2 = e;
    float wsum = sc[i1] + sc[i2];
    float w1 = sc[i1] / wsum, w2 = sc[i2] / wsum;
    atomicAdd(&ctrl[9 + i1], 1.f);
    atomicAdd(&ctrl[9 + i2], 1.f);
    int p1 = atomicAdd(&ecount[i1], 1);
    elist[i1 * NTOK + p1] = i; wlist[i1 * NTOK + p1] = w1;
    int p2 = atomicAdd(&ecount[i2], 1);
    elist[i2 * NTOK + p2] = i; wlist[i2 * NTOK + p2] = w2;
}

// ---------------------------------------------------------------- routed expert FFN (fused)
// dyn LDS: At[16][2048] bf16 + Ht[16][2048] bf16 = 128 KB (fits 320 KB WGP LDS)
__global__ void __launch_bounds__(256) k_ffn(const unsigned short* __restrict__ flat_bf,
                                             const unsigned short* __restrict__ W1t,
                                             const float* __restrict__ b1,
                                             const unsigned short* __restrict__ W2t,
                                             const float* __restrict__ b2,
                                             const int* __restrict__ ecount,
                                             const int* __restrict__ elist,
                                             const float* __restrict__ wlist,
                                             float* __restrict__ out) {
    extern __shared__ unsigned short smem[];
    unsigned short* At = smem;                 // 16 x GRU_OUT
    unsigned short* Ht = smem + 16 * GRU_OUT;  // 16 x FFN
    __shared__ int   tok_s[16];
    __shared__ float w_s[16];

    int e = blockIdx.x >> 9;
    int tile = blockIdx.x & 511;
    int cnt = ecount[e];
    if (tile * 16 >= cnt) return;
    int lane = threadIdx.x & 31, w = threadIdx.x >> 5;

    if (threadIdx.x < 16) {
        int idx = tile * 16 + threadIdx.x;
        if (idx < cnt) { tok_s[threadIdx.x] = elist[e * NTOK + idx]; w_s[threadIdx.x] = wlist[e * NTOK + idx]; }
        else           { tok_s[threadIdx.x] = elist[e * NTOK + tile * 16]; w_s[threadIdx.x] = 0.f; }
    }
    __syncthreads();

    // gather 16-token A tile straight into LDS with async copies (16B per lane per issue)
    for (int idx = threadIdx.x; idx < 16 * (GRU_OUT / 8); idx += blockDim.x) {
        int r = idx >> 8;               // GRU_OUT/8 = 256 chunks per row
        int c8 = idx & 255;             // 8 bf16 = 16 bytes per chunk
        const unsigned short* g = flat_bf + (long)tok_s[r] * GRU_OUT + c8 * 8;
        unsigned lds = (unsigned)(unsigned long long)(At + r * GRU_OUT + c8 * 8);
        async_g2l_b128(lds, g);
    }
    wait_async0();
    __syncthreads();

    // GEMM1: [16,2048] x [2048,2048] -> gelu -> Ht (bf16)
    const unsigned short* arow = At + (lane & 15) * GRU_OUT;
    for (int it = 0; it < 16; ++it) {
        int col = (w * 16 + it) * 16 + (lane & 15);
        const unsigned short* bcol = W1t + ((long)e * FFN + col) * GRU_OUT;
        float bias = b1[e * FFN + col];
        v8f c;
#pragma unroll
        for (int j = 0; j < 8; ++j) c[j] = bias;
        for (int k = 0; k < GRU_OUT; k += 32) {
            __builtin_prefetch(bcol + k + 256, 0, 0);
            c = WMMA_BF16(load_afrag(arow, k, lane), load_bfrag(bcol, k, lane), c);
        }
#pragma unroll
        for (int j = 0; j < 8; ++j) {
            float x = c[j];
            float g = 0.5f * x * (1.f + erff(x * 0.70710678f));   // exact gelu
            int row = j + ((lane & 16) ? 8 : 0);
            Ht[row * FFN + col] = f2bf(g);
        }
    }
    __syncthreads();

    // GEMM2: [16,2048] x [2048,512] -> weighted scatter-add into output
    const unsigned short* hrow = Ht + (lane & 15) * FFN;
    for (int it = 0; it < 4; ++it) {
        int col = (w * 4 + it) * 16 + (lane & 15);
        const unsigned short* bcol = W2t + ((long)e * D_MODEL + col) * FFN;
        float bias = b2[e * D_MODEL + col];
        v8f c;
#pragma unroll
        for (int j = 0; j < 8; ++j) c[j] = bias;
        for (int k = 0; k < FFN; k += 32)
            c = WMMA_BF16(load_afrag(hrow, k, lane), load_bfrag(bcol, k, lane), c);
#pragma unroll
        for (int j = 0; j < 8; ++j) {
            int row = j + ((lane & 16) ? 8 : 0);
            if (tile * 16 + row < cnt)
                atomicAdd(&out[(long)tok_s[row] * D_MODEL + col], w_s[row] * c[j]);
        }
    }
}

// ---------------------------------------------------------------- aux loss finalize
__global__ void k_fin(const float* ctrl, float* out) {
    if (threadIdx.x == 0 && blockIdx.x == 0) {
        float n = (float)NTOK;
        float zl = 0.001f * ctrl[0] / n;
        float bl = 0.f;
        for (int e = 0; e < NEXP; ++e) bl += (ctrl[9 + e] / n) * (ctrl[1 + e] / n);
        out[OUTN] = 0.01f * (float)NEXP * bl + zl;
    }
}

// ----------------------------------------------------------------
extern "C" void kernel_launch(void* const* d_in, const int* in_sizes, int n_in,
                              void* d_out, int out_size, void* d_ws, size_t ws_size,
                              hipStream_t stream) {
    const float* x   = (const float*)d_in[0];
    const float* Wih = (const float*)d_in[1];
    const float* Whh = (const float*)d_in[2];
    const float* bih = (const float*)d_in[3];
    const float* bhh = (const float*)d_in[4];
    const float* gW  = (const float*)d_in[5];
    const float* gb  = (const float*)d_in[6];
    const float* W1  = (const float*)d_in[7];
    const float* b1  = (const float*)d_in[8];
    const float* W2  = (const float*)d_in[9];
    const float* b2  = (const float*)d_in[10];
    float* out = (float*)d_out;

    char* p = (char*)d_ws;
    auto alloc = [&](size_t bytes) { void* r = (void*)p; p += (bytes + 255) & ~(size_t)255; return r; };
    unsigned short* x_bf    = (unsigned short*)alloc((size_t)NTOK * D_MODEL * 2);
    unsigned short* Wih_bf  = (unsigned short*)alloc((size_t)2 * 3 * H * D_MODEL * 2);
    unsigned short* Whh_bf  = (unsigned short*)alloc((size_t)2 * 3 * H * H * 2);
    unsigned short* W1t_bf  = (unsigned short*)alloc((size_t)NEXP * FFN * GRU_OUT * 2);
    unsigned short* W2t_bf  = (unsigned short*)alloc((size_t)NEXP * D_MODEL * FFN * 2);
    unsigned short* xg_bf   = (unsigned short*)alloc((size_t)2 * TLEN * BATCH * 3 * H * 2);
    unsigned short* flat_bf = (unsigned short*)alloc((size_t)NTOK * GRU_OUT * 2);
    int*   elist = (int*)alloc((size_t)NEXP * NTOK * 4);
    float* wlist = (float*)alloc((size_t)NEXP * NTOK * 4);
    float* ctrl  = (float*)alloc(64 * 4);
    int*   ecount = (int*)(ctrl + 32);

    k_init<<<1024, 256, 0, stream>>>(out, ctrl);
    k_f2bf<<<1024, 256, 0, stream>>>(x,   x_bf,   (long)NTOK * D_MODEL);
    k_f2bf<<<1024, 256, 0, stream>>>(Wih, Wih_bf, (long)2 * 3 * H * D_MODEL);
    k_f2bf<<<1024, 256, 0, stream>>>(Whh, Whh_bf, (long)2 * 3 * H * H);
    k_t2bf<<<4096, 256, 0, stream>>>(W1, W1t_bf, NEXP, GRU_OUT, FFN);
    k_t2bf<<<4096, 256, 0, stream>>>(W2, W2t_bf, NEXP, FFN, D_MODEL);
    k_xproj<<<24576, 256, 0, stream>>>(x_bf, Wih_bf, bih, xg_bf);
    k_gru<<<2, 1024, 0, stream>>>(xg_bf, Whh_bf, bhh, flat_bf);
    k_gate<<<NTOK / 256, 256, 0, stream>>>(flat_bf, gW, gb, ctrl, ecount, elist, wlist);
    k_ffn<<<NEXP * 512, 256, 2 * 16 * 2048 * 2, stream>>>(flat_bf, W1t_bf, b1, W2t_bf, b2,
                                                          ecount, elist, wlist, out);
    k_fin<<<1, 1, 0, stream>>>(ctrl, out);
}
